// GaussianImage_Cholesky_9371618639957
// MI455X (gfx1250) — matrix operand order repaired and checked by reading, not compiled, not run on hardware
//
#include <hip/hip_runtime.h>
#include <hip/hip_bf16.h>

#define IMG_H 256
#define IMG_W 256
#define NGAUSS 1024
#define ALPHA_THRESH (1.0f / 255.0f)

typedef __attribute__((ext_vector_type(16))) _Float16 v16h;
typedef __attribute__((ext_vector_type(8)))  float    v8f;

// ---------------------------------------------------------------------------
// Kernel 1: projection. 1024 Gaussians -> packed params in d_ws.
// Layout per Gaussian (2 x float4 = 32 B):
//   q0 = { cx, cy, conic_a, conic_b }
//   q1 = { conic_c, opacity, pack(f0,f1) as 2xf16, pack(f2,0) as 2xf16 }
// ---------------------------------------------------------------------------
__global__ void gsplat_project_kernel(const float* __restrict__ xyz,
                                      const float* __restrict__ chol,
                                      const float* __restrict__ feat,
                                      const float* __restrict__ opac,
                                      float4* __restrict__ proj) {
    int i = blockIdx.x * blockDim.x + threadIdx.x;
    if (i >= NGAUSS) return;
    float mx = tanhf(xyz[i * 2 + 0]);
    float my = tanhf(xyz[i * 2 + 1]);
    float l1 = chol[i * 3 + 0] + 0.5f;
    float l2 = chol[i * 3 + 1];
    float l3 = chol[i * 3 + 2] + 0.5f;
    float cx = 0.5f * (float)IMG_W * (mx + 1.0f);
    float cy = 0.5f * (float)IMG_H * (my + 1.0f);
    float a = l1 * l1;
    float b = l1 * l2;
    float c = l2 * l2 + l3 * l3;
    float inv_det = 1.0f / (a * c - b * b);
    float ca = c * inv_det;
    float cb = -b * inv_det;
    float cc = a * inv_det;
    union { float f; _Float16 h[2]; } p01, p2;
    p01.h[0] = (_Float16)feat[i * 3 + 0];
    p01.h[1] = (_Float16)feat[i * 3 + 1];
    p2.h[0]  = (_Float16)feat[i * 3 + 2];
    p2.h[1]  = (_Float16)0.0f;
    proj[i * 2 + 0] = make_float4(cx, cy, ca, cb);
    proj[i * 2 + 1] = make_float4(cc, opac[i], p01.f, p2.f);
}

// ---------------------------------------------------------------------------
// Kernel 2: render. One wave32 per 16-pixel tile; K-loop over 32-Gaussian
// chunks using v_wmma_f32_16x16x32_f16 for the w x features einsum.
// All projected params live in 32KB LDS (WGP has 320KB).
// ---------------------------------------------------------------------------
__global__ __launch_bounds__(256) void gsplat_render_kernel(
        const float4* __restrict__ proj, float* __restrict__ out) {
    __shared__ float4 sh[NGAUSS * 2];  // 32 KB

    const int tid = threadIdx.x;
    // Cooperative LDS fill: 2048 float4 / 256 threads = 8 x ds-friendly b128.
    #pragma unroll
    for (int i = 0; i < 8; ++i) sh[tid + i * 256] = proj[tid + i * 256];
    __syncthreads();

    const int wave = tid >> 5;
    const int lane = tid & 31;
    const int tile = blockIdx.x * 8 + wave;        // 4096 tiles of 16 pixels
    const int m    = lane & 15;                    // pixel row within tile (M)
    const int khi  = (lane >> 4) << 3;             // +8 K/M offset for hi lanes
    const int p0   = tile * 16;                    // flat pixel base (row-aligned)
    const float px = (float)((p0 + m) & (IMG_W - 1)) + 0.5f;
    const float py = (float)((p0 + m) >> 8) + 0.5f;  // W == 256

    v8f acc = {};  // 16x16 f32 accumulator (cols = channels, rows = pixels)

    for (int chunk = 0; chunk < NGAUSS / 32; ++chunk) {
        const int gbase = chunk * 32;

        // ---- A fragment: w[m, k] packed per ISA 16-bit A 16x32 layout ----
        v16h afrag;
        #pragma unroll
        for (int i = 0; i < 16; ++i) {
            const int k = i + ((i >> 3) << 3) + khi;   // element -> K mapping
            const int g = gbase + k;
            const float4 q0 = sh[g * 2 + 0];
            const float4 q1 = sh[g * 2 + 1];
            const float dx = q0.x - px;
            const float dy = q0.y - py;
            const float sigma = 0.5f * (q0.z * dx * dx + q1.x * dy * dy)
                              + q0.w * dx * dy;
            float alpha = fminf(0.999f, q1.y * __expf(-sigma));
            float w = (sigma >= 0.0f && alpha >= ALPHA_THRESH) ? alpha : 0.0f;
            afrag[i] = (_Float16)w;
        }

        // ---- B fragment: lane column n = lane%16 (channel), elements over K (same order as A) ----
        v16h bfrag = {};
        if (m < 3) {
            #pragma unroll
            for (int i = 0; i < 16; ++i) {
                const int k = i + ((i >> 3) << 3) + khi;
                const float4 qb = sh[(gbase + k) * 2 + 1];
                union { float f; _Float16 h[2]; } u0;
                u0.f = (m < 2) ? qb.z : qb.w;
                bfrag[i] = u0.h[m & 1];
            }
        }

        // D = A x B + C : v_wmma_f32_16x16x32_f16
        acc = __builtin_amdgcn_wmma_f32_16x16x32_f16(
            /*neg_a=*/false, afrag, /*neg_b=*/false, bfrag,
            /*c_mod=*/(short)0, acc, /*reuse_a=*/false, /*reuse_b=*/false);
    }

    // ---- Store: C/D layout cols(N)=lane%16, row M = vgpr r (+8 hi lanes) ----
    const int n = lane & 15;
    if (n < 3) {
        #pragma unroll
        for (int r = 0; r < 8; ++r) {
            const int p = p0 + r + khi;
            float v = acc[r];
            v = fminf(fmaxf(v, 0.0f), 1.0f);
            out[n * (IMG_H * IMG_W) + p] = v;
        }
    }
}

extern "C" void kernel_launch(void* const* d_in, const int* in_sizes, int n_in,
                              void* d_out, int out_size, void* d_ws, size_t ws_size,
                              hipStream_t stream) {
    const float* xyz  = (const float*)d_in[0];   // (1024, 2) f32
    const float* chol = (const float*)d_in[1];   // (1024, 3) f32
    const float* feat = (const float*)d_in[2];   // (1024, 3) f32
    const float* opac = (const float*)d_in[3];   // (1024, 1) f32
    float*  out  = (float*)d_out;                // (1, 3, 256, 256) f32
    float4* proj = (float4*)d_ws;                // 1024 x 32 B = 32 KB scratch

    gsplat_project_kernel<<<NGAUSS / 256, 256, 0, stream>>>(xyz, chol, feat, opac, proj);
    // 4096 tiles / 8 waves-per-block = 512 blocks of 256 threads (8 wave32s).
    gsplat_render_kernel<<<512, 256, 0, stream>>>(proj, out);
}
